// AnchorSelfAttention1d_23390391894154
// MI455X (gfx1250) — matrix operand order repaired and checked by reading, not compiled, not run on hardware
//
#include <hip/hip_runtime.h>
#include <hip/hip_bf16.h>
#include <stddef.h>

typedef __bf16 bf16_t;
typedef __attribute__((ext_vector_type(16))) __bf16 v16bf;
typedef __attribute__((ext_vector_type(8)))  __bf16 v8bf;
typedef __attribute__((ext_vector_type(8)))  float  v8f;

#define DIMD 512
#define LENL 8192
#define NB   8
#define PREAL 100
#define PPAD 112   // P padded to 16
#define CPAD 128   // P padded to 32 (WMMA K granularity)

// ---------------------------------------------------------------------------
// D = A(16x32 bf16) * B(32x16 bf16) + C(16x16 f32), wave32 WMMA
// ---------------------------------------------------------------------------
__device__ __forceinline__ v8f wmma_bf16(v16bf a, v16bf b, v8f c) {
    return __builtin_amdgcn_wmma_f32_16x16x32_bf16(
        /*neg_a=*/false, a, /*neg_b=*/false, b,
        /*c_mod=*/(short)0, c, /*reuse_a=*/false, /*reuse_b=*/false);
}

// A fragment (16x32) from a row-major bf16 matrix, rows contiguous.
// base points at element (row 0, k 0) of the tile; ld = row stride (elems).
// Layout: lanes 0-15 row M=lane, K 0..15; lanes 16-31 K 16..31.
__device__ __forceinline__ v16bf frag_a_rowmajor(const bf16_t* base, int ld, int lane) {
    int m  = lane & 15;
    int kb = (lane & 16) ? 16 : 0;
    const bf16_t* row = base + (size_t)m * ld + kb;
    v8bf lo = *(const v8bf*)(row);
    v8bf hi = *(const v8bf*)(row + 8);
    v16bf f;
#pragma unroll
    for (int j = 0; j < 8; ++j) { f[j] = lo[j]; f[j + 8] = hi[j]; }
    return f;
}

// A fragment (16x32) built from f32 data stored K-major (element (m,kk) at
// base[kk*ldk + m]) — used for channels-first x (implicit transpose + cvt).
__device__ __forceinline__ v16bf frag_a_f32_kmajor(const float* base, int ldk, int lane) {
    int m  = lane & 15;
    int kb = (lane & 16) ? 16 : 0;
    v16bf f;
#pragma unroll
    for (int j = 0; j < 8; ++j) {
        f[j]     = (bf16_t)base[(size_t)(kb + j) * ldk + m];
        f[j + 8] = (bf16_t)base[(size_t)(kb + 8 + j) * ldk + m];
    }
    return f;
}

// B fragment (32x16) where B[k][n] = M[n][k], M row-major (multiply by M^T).
// Column n is contiguous in memory -> two 16B loads per lane.
// Layout: lanes 0-15 col N=lane K 0..15; lanes 16-31 K 16..31.
__device__ __forceinline__ v16bf frag_b_transposed(const bf16_t* base, int ld, int lane) {
    int n    = lane & 15;
    int koff = (lane & 16) ? 16 : 0;
    const bf16_t* col = base + (size_t)n * ld + koff;
    v8bf lo = *(const v8bf*)(col);
    v8bf hi = *(const v8bf*)(col + 8);
    v16bf f;
#pragma unroll
    for (int j = 0; j < 8; ++j) { f[j] = lo[j]; f[j + 8] = hi[j]; }
    return f;
}

// B fragment (32x16) where B[k][n] = M[k*ld + n], M row-major [K x N] (strided).
__device__ __forceinline__ v16bf frag_b_strided(const bf16_t* base, int ld, int lane) {
    int n    = lane & 15;
    int koff = (lane & 16) ? 16 : 0;
    v16bf f;
#pragma unroll
    for (int j = 0; j < 16; ++j) f[j] = base[(size_t)(koff + j) * ld + n];
    return f;
}

// ---------------------------------------------------------------------------
// 0) fp32 weights -> bf16
// ---------------------------------------------------------------------------
__global__ void cvt_w_kernel(const float* __restrict__ Wq, const float* __restrict__ Wk,
                             const float* __restrict__ Wv,
                             bf16_t* __restrict__ Wqb, bf16_t* __restrict__ Wkb,
                             bf16_t* __restrict__ Wvb) {
    int i = blockIdx.x * 256 + threadIdx.x;
    if (i < DIMD * DIMD) {
        Wqb[i] = (bf16_t)Wq[i];
        Wkb[i] = (bf16_t)Wk[i];
        Wvb[i] = (bf16_t)Wv[i];
    }
}

// ---------------------------------------------------------------------------
// 1) fused projections: q/k/v = x^T W^T + b  -> bf16 [B, L, D]
//    grid = B * L/16 blocks; wave w owns output cols [64w, 64w+64)
// ---------------------------------------------------------------------------
__global__ __launch_bounds__(256)
void proj_qkv_kernel(const float* __restrict__ x,
                     const bf16_t* __restrict__ Wqb, const bf16_t* __restrict__ Wkb,
                     const bf16_t* __restrict__ Wvb,
                     const float* __restrict__ bq, const float* __restrict__ bk,
                     const float* __restrict__ bv,
                     bf16_t* __restrict__ Qb, bf16_t* __restrict__ Kb,
                     bf16_t* __restrict__ Vb) {
    int blk  = blockIdx.x;
    int b    = blk / (LENL / 16);
    int l0   = (blk % (LENL / 16)) * 16;
    int wave = threadIdx.x >> 5;
    int lane = threadIdx.x & 31;
    int n0   = wave * 64;

    const float* xbase = x + (size_t)b * DIMD * LENL + l0;  // (m,d) at xbase[d*L + m]

    v8f accq[4] = {}, acck[4] = {}, accv[4] = {};
    for (int k0 = 0; k0 < DIMD; k0 += 32) {
        v16bf af = frag_a_f32_kmajor(xbase + (size_t)k0 * LENL, LENL, lane);
#pragma unroll
        for (int t = 0; t < 4; ++t) {
            int n = n0 + t * 16;
            v16bf bqf = frag_b_transposed(Wqb + (size_t)n * DIMD + k0, DIMD, lane);
            accq[t] = wmma_bf16(af, bqf, accq[t]);
            v16bf bkf = frag_b_transposed(Wkb + (size_t)n * DIMD + k0, DIMD, lane);
            acck[t] = wmma_bf16(af, bkf, acck[t]);
            v16bf bvf = frag_b_transposed(Wvb + (size_t)n * DIMD + k0, DIMD, lane);
            accv[t] = wmma_bf16(af, bvf, accv[t]);
        }
    }

    int nl = lane & 15;
    int mb = (lane & 16) ? 8 : 0;
#pragma unroll
    for (int t = 0; t < 4; ++t) {
        int n = n0 + t * 16 + nl;
        float bqv = bq[n], bkv = bk[n], bvv = bv[n];
        size_t ob = (size_t)b * LENL * DIMD + (size_t)l0 * DIMD + n;
#pragma unroll
        for (int j = 0; j < 8; ++j) {
            size_t o = ob + (size_t)(mb + j) * DIMD;
            Qb[o] = (bf16_t)(accq[t][j] + bqv);
            Kb[o] = (bf16_t)(acck[t][j] + bkv);
            Vb[o] = (bf16_t)(accv[t][j] + bvv);
        }
    }
}

// ---------------------------------------------------------------------------
// 2) adaptive-avg-pool anchors from k -> bf16 [B, PPAD, D] (rows >= P zeroed)
// ---------------------------------------------------------------------------
__global__ __launch_bounds__(256)
void anchors_kernel(const bf16_t* __restrict__ Kb, bf16_t* __restrict__ Ab) {
    int b = blockIdx.x / PPAD;
    int p = blockIdx.x % PPAD;
    for (int d = threadIdx.x; d < DIMD; d += 256) {
        float s = 0.f;
        if (p < PREAL) {
            int start = (p * LENL) / PREAL;
            int end   = ((p + 1) * LENL + PREAL - 1) / PREAL;
            float inv = 1.f / (float)(end - start);
            for (int l = start; l < end; ++l)
                s += (float)Kb[(size_t)b * LENL * DIMD + (size_t)l * DIMD + d];
            s *= inv;
        }
        Ab[(size_t)b * PPAD * DIMD + (size_t)p * DIMD + d] = (bf16_t)s;
    }
}

// ---------------------------------------------------------------------------
// 3a) stage-1 scores: S[b,p,l] = a[b,p,:] . k[b,l,:]   (f32 [B, PPAD, L])
//     grid (L/512, PPAD/16, B); wave owns 64 l-columns
// ---------------------------------------------------------------------------
__global__ __launch_bounds__(256)
void scores1_kernel(const bf16_t* __restrict__ Ab, const bf16_t* __restrict__ Kb,
                    float* __restrict__ S) {
    int b    = blockIdx.z;
    int p0   = blockIdx.y * 16;
    int wave = threadIdx.x >> 5;
    int lane = threadIdx.x & 31;
    int l0   = blockIdx.x * 512 + wave * 64;

    const bf16_t* abase = Ab + (size_t)b * PPAD * DIMD + (size_t)p0 * DIMD;
    const bf16_t* kbase = Kb + (size_t)b * LENL * DIMD;

    v8f acc[4] = {};
    for (int k0 = 0; k0 < DIMD; k0 += 32) {
        v16bf af = frag_a_rowmajor(abase + k0, DIMD, lane);
#pragma unroll
        for (int t = 0; t < 4; ++t) {
            v16bf bf_ = frag_b_transposed(kbase + (size_t)(l0 + t * 16) * DIMD + k0, DIMD, lane);
            acc[t] = wmma_bf16(af, bf_, acc[t]);
        }
    }
    int nl = lane & 15, mb = (lane & 16) ? 8 : 0;
#pragma unroll
    for (int t = 0; t < 4; ++t) {
        size_t base = (size_t)b * PPAD * LENL + (size_t)p0 * LENL + (l0 + t * 16 + nl);
#pragma unroll
        for (int j = 0; j < 8; ++j)
            S[base + (size_t)(mb + j) * LENL] = acc[t][j];
    }
}

// ---------------------------------------------------------------------------
// 3b) softmax over L per (b,p) row -> bf16 probs [B, PPAD, L]
// ---------------------------------------------------------------------------
__global__ __launch_bounds__(256)
void softmax1_kernel(const float* __restrict__ S, bf16_t* __restrict__ P1) {
    int b = blockIdx.x / PPAD, p = blockIdx.x % PPAD;
    const float* row  = S  + (size_t)b * PPAD * LENL + (size_t)p * LENL;
    bf16_t*      orow = P1 + (size_t)b * PPAD * LENL + (size_t)p * LENL;
    __shared__ float red[256];

    float m = -1e30f;
    for (int i = threadIdx.x; i < LENL; i += 256) m = fmaxf(m, row[i]);
    red[threadIdx.x] = m;
    __syncthreads();
    for (int s = 128; s > 0; s >>= 1) {
        if (threadIdx.x < s) red[threadIdx.x] = fmaxf(red[threadIdx.x], red[threadIdx.x + s]);
        __syncthreads();
    }
    m = red[0];
    __syncthreads();

    float sum = 0.f;
    for (int i = threadIdx.x; i < LENL; i += 256) sum += __expf(row[i] - m);
    red[threadIdx.x] = sum;
    __syncthreads();
    for (int s = 128; s > 0; s >>= 1) {
        if (threadIdx.x < s) red[threadIdx.x] += red[threadIdx.x + s];
        __syncthreads();
    }
    float inv = 1.f / red[0];
    for (int i = threadIdx.x; i < LENL; i += 256)
        orow[i] = (bf16_t)(__expf(row[i] - m) * inv);
}

// ---------------------------------------------------------------------------
// 3c) context: c[b,p,d] = sum_l s1[p,l] v[l,d] -> bf16 [B, CPAD, D]
//     grid = B * CPAD/16; wave owns 64 d-columns. Rows >= PPAD zeroed.
// ---------------------------------------------------------------------------
__global__ __launch_bounds__(256)
void ctx1_kernel(const bf16_t* __restrict__ P1, const bf16_t* __restrict__ Vb,
                 bf16_t* __restrict__ Cb) {
    int b    = blockIdx.x >> 3;
    int p0   = (blockIdx.x & 7) * 16;
    int wave = threadIdx.x >> 5;
    int lane = threadIdx.x & 31;
    int d0   = wave * 64;
    int nl   = lane & 15, mb = (lane & 16) ? 8 : 0;

    if (p0 >= PPAD) {  // pad rows 112..127 -> 0
#pragma unroll
        for (int t = 0; t < 4; ++t)
#pragma unroll
            for (int j = 0; j < 8; ++j)
                Cb[(size_t)b * CPAD * DIMD + (size_t)(p0 + mb + j) * DIMD + d0 + t * 16 + nl] =
                    (bf16_t)0.f;
        return;
    }

    const bf16_t* pbase = P1 + (size_t)b * PPAD * LENL + (size_t)p0 * LENL;
    const bf16_t* vbase = Vb + (size_t)b * LENL * DIMD;

    v8f acc[4] = {};
    for (int l0 = 0; l0 < LENL; l0 += 32) {
        v16bf af = frag_a_rowmajor(pbase + l0, LENL, lane);
#pragma unroll
        for (int t = 0; t < 4; ++t) {
            v16bf bf_ = frag_b_strided(vbase + (size_t)l0 * DIMD + d0 + t * 16, DIMD, lane);
            acc[t] = wmma_bf16(af, bf_, acc[t]);
        }
    }
#pragma unroll
    for (int t = 0; t < 4; ++t)
#pragma unroll
        for (int j = 0; j < 8; ++j)
            Cb[(size_t)b * CPAD * DIMD + (size_t)(p0 + mb + j) * DIMD + d0 + t * 16 + nl] =
                (bf16_t)acc[t][j];
}

// ---------------------------------------------------------------------------
// 4a) stage-2 scores: S[b,l,p] = q[b,l,:] . a[b,p,:]   (f32 [B, L, PPAD])
//     grid = B * L/128; wave owns 16 l-rows, all 7 p-tiles
// ---------------------------------------------------------------------------
__global__ __launch_bounds__(256)
void scores2_kernel(const bf16_t* __restrict__ Qb, const bf16_t* __restrict__ Ab,
                    float* __restrict__ S) {
    int b    = blockIdx.x >> 6;
    int l0   = (blockIdx.x & 63) * 128 + (threadIdx.x >> 5) * 16;
    int lane = threadIdx.x & 31;

    const bf16_t* qbase = Qb + (size_t)b * LENL * DIMD + (size_t)l0 * DIMD;
    const bf16_t* abase = Ab + (size_t)b * PPAD * DIMD;

    v8f acc[7] = {};
    for (int k0 = 0; k0 < DIMD; k0 += 32) {
        v16bf af = frag_a_rowmajor(qbase + k0, DIMD, lane);
#pragma unroll
        for (int t = 0; t < 7; ++t) {
            v16bf bf_ = frag_b_transposed(abase + (size_t)(t * 16) * DIMD + k0, DIMD, lane);
            acc[t] = wmma_bf16(af, bf_, acc[t]);
        }
    }
    int nl = lane & 15, mb = (lane & 16) ? 8 : 0;
#pragma unroll
    for (int t = 0; t < 7; ++t) {
        size_t base = (size_t)b * LENL * PPAD + (size_t)l0 * PPAD + t * 16 + nl;
#pragma unroll
        for (int j = 0; j < 8; ++j)
            S[base + (size_t)(mb + j) * PPAD] = acc[t][j];
    }
}

// ---------------------------------------------------------------------------
// 4b) softmax over P (mask p>=100) -> bf16 probs [B, L, CPAD], cols>=100 = 0
//     one wave per row; 8 rows per block
// ---------------------------------------------------------------------------
__global__ __launch_bounds__(256)
void softmax2_kernel(const float* __restrict__ S, bf16_t* __restrict__ P2) {
    int row  = blockIdx.x * 8 + (threadIdx.x >> 5);  // flat (b*L + l)
    int lane = threadIdx.x & 31;
    const float* r = S  + (size_t)row * PPAD;
    bf16_t*      o = P2 + (size_t)row * CPAD;

    float v[4];
    float m = -1e30f;
#pragma unroll
    for (int j = 0; j < 4; ++j) {
        int c = lane + j * 32;
        v[j] = (c < PREAL) ? r[c] : -1e30f;
        m = fmaxf(m, v[j]);
    }
#pragma unroll
    for (int off = 16; off > 0; off >>= 1) m = fmaxf(m, __shfl_xor(m, off, 32));

    float s = 0.f;
#pragma unroll
    for (int j = 0; j < 4; ++j) {
        v[j] = (lane + j * 32 < PREAL) ? __expf(v[j] - m) : 0.f;
        s += v[j];
    }
#pragma unroll
    for (int off = 16; off > 0; off >>= 1) s += __shfl_xor(s, off, 32);

    float inv = 1.f / s;
#pragma unroll
    for (int j = 0; j < 4; ++j) o[lane + j * 32] = (bf16_t)(v[j] * inv);
}

// ---------------------------------------------------------------------------
// 5) r[b,l,d] = sum_p s2[l,p] c[p,d]; scatter to channels-first f32 out [B,D,L]
//    grid = B * L/16; wave owns 64 d-columns; K = CPAD = 128
// ---------------------------------------------------------------------------
__global__ __launch_bounds__(256)
void final_kernel(const bf16_t* __restrict__ P2, const bf16_t* __restrict__ Cb,
                  float* __restrict__ out) {
    int b    = blockIdx.x >> 9;
    int l0   = (blockIdx.x & 511) * 16;
    int wave = threadIdx.x >> 5;
    int lane = threadIdx.x & 31;
    int d0   = wave * 64;

    const bf16_t* pbase = P2 + ((size_t)b * LENL + l0) * CPAD;
    const bf16_t* cbase = Cb + (size_t)b * CPAD * DIMD;

    v8f acc[4] = {};
#pragma unroll
    for (int k0 = 0; k0 < CPAD; k0 += 32) {
        v16bf af = frag_a_rowmajor(pbase + k0, CPAD, lane);
#pragma unroll
        for (int t = 0; t < 4; ++t) {
            v16bf bf_ = frag_b_strided(cbase + (size_t)k0 * DIMD + d0 + t * 16, DIMD, lane);
            acc[t] = wmma_bf16(af, bf_, acc[t]);
        }
    }
    int nl = lane & 15, mb = (lane & 16) ? 8 : 0;
#pragma unroll
    for (int t = 0; t < 4; ++t) {
        int d = d0 + t * 16 + nl;
        size_t base = (size_t)b * DIMD * LENL + (size_t)d * LENL + l0;
#pragma unroll
        for (int j = 0; j < 8; ++j)
            out[base + mb + j] = acc[t][j];
    }
}

// ---------------------------------------------------------------------------
// host launcher
// ---------------------------------------------------------------------------
extern "C" void kernel_launch(void* const* d_in, const int* in_sizes, int n_in,
                              void* d_out, int out_size, void* d_ws, size_t ws_size,
                              hipStream_t stream) {
    const float* x  = (const float*)d_in[0];
    const float* Wq = (const float*)d_in[1];
    const float* bq = (const float*)d_in[2];
    const float* Wk = (const float*)d_in[3];
    const float* bk = (const float*)d_in[4];
    const float* Wv = (const float*)d_in[5];
    const float* bv = (const float*)d_in[6];
    float* out = (float*)d_out;

    char* ws = (char*)d_ws;
    const size_t SZ_QKV = (size_t)NB * LENL * DIMD * sizeof(bf16_t);  // 64 MB each
    const size_t SZ_A   = (size_t)NB * PPAD * DIMD * sizeof(bf16_t);
    const size_t SZ_C   = (size_t)NB * CPAD * DIMD * sizeof(bf16_t);
    const size_t SZ_S   = (size_t)NB * PPAD * LENL * sizeof(float);   // reused for both score mats
    const size_t SZ_P1  = (size_t)NB * PPAD * LENL * sizeof(bf16_t);
    const size_t SZ_P2  = (size_t)NB * LENL * CPAD * sizeof(bf16_t);
    const size_t SZ_W   = (size_t)DIMD * DIMD * sizeof(bf16_t);

    size_t off = 0;
    bf16_t* Qb  = (bf16_t*)(ws + off); off += SZ_QKV;
    bf16_t* Kb  = (bf16_t*)(ws + off); off += SZ_QKV;
    bf16_t* Vb  = (bf16_t*)(ws + off); off += SZ_QKV;
    bf16_t* Ab  = (bf16_t*)(ws + off); off += SZ_A;
    bf16_t* Cb  = (bf16_t*)(ws + off); off += SZ_C;
    float*  Sc  = (float*) (ws + off); off += SZ_S;
    bf16_t* P1  = (bf16_t*)(ws + off); off += SZ_P1;
    bf16_t* P2  = (bf16_t*)(ws + off); off += SZ_P2;
    bf16_t* Wqb = (bf16_t*)(ws + off); off += SZ_W;
    bf16_t* Wkb = (bf16_t*)(ws + off); off += SZ_W;
    bf16_t* Wvb = (bf16_t*)(ws + off); off += SZ_W;
    (void)in_sizes; (void)n_in; (void)out_size; (void)ws_size;

    // 0) weights -> bf16
    cvt_w_kernel<<<(DIMD * DIMD + 255) / 256, 256, 0, stream>>>(Wq, Wk, Wv, Wqb, Wkb, Wvb);
    // 1) projections
    proj_qkv_kernel<<<NB * (LENL / 16), 256, 0, stream>>>(x, Wqb, Wkb, Wvb, bq, bk, bv,
                                                          Qb, Kb, Vb);
    // 2) anchors
    anchors_kernel<<<NB * PPAD, 256, 0, stream>>>(Kb, Ab);
    // 3) stage 1
    scores1_kernel<<<dim3(LENL / 512, PPAD / 16, NB), 256, 0, stream>>>(Ab, Kb, Sc);
    softmax1_kernel<<<NB * PPAD, 256, 0, stream>>>(Sc, P1);
    ctx1_kernel<<<NB * (CPAD / 16), 256, 0, stream>>>(P1, Vb, Cb);
    // 4) stage 2
    scores2_kernel<<<NB * (LENL / 128), 256, 0, stream>>>(Qb, Ab, Sc);
    softmax2_kernel<<<NB * LENL / 8, 256, 0, stream>>>(Sc, P2);
    // 5) output
    final_kernel<<<NB * (LENL / 16), 256, 0, stream>>>(P2, Cb, out);
}
